// Cluster_Level_GCN_63745904607637
// MI455X (gfx1250) — matrix operand order repaired and checked by reading, not compiled, not run on hardware
//
#include <hip/hip_runtime.h>
#include <hip/hip_bf16.h>
#include <math.h>

// Problem constants (from reference)
#define N_PTS 65536
#define NUM_CL 8192
#define BB 256
#define KNN 64
#define DD 2048
#define NHID 512
#define DOUT 256
#define NCLS 2

typedef __attribute__((ext_vector_type(16))) __bf16 v16bf;
typedef __attribute__((ext_vector_type(8)))  __bf16 v8bf;
typedef __attribute__((ext_vector_type(8)))  float  v8f;

// Async 16-byte global -> LDS copy (per lane). Tracked by ASYNCcnt.
// LDS operand: low 32 bits of the generic shared-aperture address are the
// wave-relative LDS byte offset (ISA 10.2 aperture mapping).
__device__ __forceinline__ void async_copy16(const void* g, void* l) {
    unsigned loff = (unsigned)(uintptr_t)l;
    asm volatile("global_load_async_to_lds_b128 %0, %1, off"
                 :: "v"(loff), "v"(g) : "memory");
}
__device__ __forceinline__ void wait_async0() {
    asm volatile("s_wait_asynccnt 0" ::: "memory");
}

// ---------------------------------------------------------------------------
// Utility: zero f32 buffer
// ---------------------------------------------------------------------------
__global__ void zero_f32(float* __restrict__ p, long n) {
    long i = (long)blockIdx.x * blockDim.x + threadIdx.x;
    long stride = (long)gridDim.x * blockDim.x;
    for (; i < n; i += stride) p[i] = 0.0f;
}

// ---------------------------------------------------------------------------
// Utility: f32 -> bf16 convert
// ---------------------------------------------------------------------------
__global__ void f32_to_bf16(const float* __restrict__ s, __bf16* __restrict__ d,
                            long n) {
    long i = (long)blockIdx.x * blockDim.x + threadIdx.x;
    long stride = (long)gridDim.x * blockDim.x;
    for (; i < n; i += stride) d[i] = (__bf16)s[i];
}

// ---------------------------------------------------------------------------
// segment_sum: one block per feature row; f32 atomics land in L2 (64MB buf)
// ---------------------------------------------------------------------------
__global__ __launch_bounds__(256) void segsum_kernel(
    const float* __restrict__ features, const int* __restrict__ labels,
    float* __restrict__ clu_sum, float* __restrict__ nums) {
    int n = blockIdx.x;
    int lab = labels[n];
    const float* f = features + (size_t)n * DD;
    float* dst = clu_sum + (size_t)lab * DD;
    for (int d = threadIdx.x; d < DD; d += 256) atomicAdd(&dst[d], f[d]);
    if (threadIdx.x == 0) atomicAdd(&nums[lab], 1.0f);
}

// ---------------------------------------------------------------------------
// clu_sum /= max(nums,1); simm = sum(clu_sum^2) per cluster
// ---------------------------------------------------------------------------
__global__ __launch_bounds__(256) void finalize_kernel(
    float* __restrict__ clu_sum, const float* __restrict__ nums,
    float* __restrict__ simm) {
    __shared__ float red[256];
    int c = blockIdx.x;
    float n = nums[c];
    float inv = 1.0f / (n > 0.0f ? n : 1.0f);
    float* p = clu_sum + (size_t)c * DD;
    float ss = 0.0f;
    for (int d = threadIdx.x; d < DD; d += 256) {
        float v = p[d] * inv;
        p[d] = v;
        ss += v * v;
    }
    red[threadIdx.x] = ss;
    __syncthreads();
    for (int s = 128; s; s >>= 1) {
        if (threadIdx.x < s) red[threadIdx.x] += red[threadIdx.x + s];
        __syncthreads();
    }
    if (threadIdx.x == 0) simm[c] = red[0];
}

// ---------------------------------------------------------------------------
// Build clu_feat [B,K,D] as bf16
// ---------------------------------------------------------------------------
__global__ __launch_bounds__(256) void build_feat_kernel(
    const float* __restrict__ features, const float* __restrict__ clu_sum,
    const int* __restrict__ labels, const int* __restrict__ indexes,
    const int* __restrict__ knn, __bf16* __restrict__ Xb) {
    int bk = blockIdx.x;
    int b = bk >> 6, k = bk & 63;
    const float* src;
    if (k == 0) {
        src = features + (size_t)indexes[b] * DD;
    } else {
        int nb = knn[(size_t)b * KNN + k];
        int lab = labels[nb];
        src = clu_sum + (size_t)lab * DD;
    }
    __bf16* dst = Xb + ((size_t)b * KNN + k) * DD;
    for (int d = threadIdx.x; d < DD; d += 256) dst[d] = (__bf16)src[d];
}

// ---------------------------------------------------------------------------
// Transpose per batch: Xt[b][d][k] = Xb[b][k][d]
// ---------------------------------------------------------------------------
__global__ __launch_bounds__(256) void transpose_kernel(
    const __bf16* __restrict__ Xb, __bf16* __restrict__ Xt) {
    __shared__ __align__(16) __bf16 t[32][72];
    int b = blockIdx.y;
    int d0 = blockIdx.x * 32;
    const __bf16* src = Xb + (size_t)b * KNN * DD;
    __bf16* dst = Xt + (size_t)b * DD * KNN;
    int k = threadIdx.x >> 2;
    int j = threadIdx.x & 3;
    uint4 v = *(const uint4*)(src + (size_t)k * DD + d0 + j * 8);
    const __bf16* pv = (const __bf16*)&v;
#pragma unroll
    for (int e = 0; e < 8; ++e) t[j * 8 + e][k] = pv[e];
    __syncthreads();
    int d = threadIdx.x >> 3;
    int k0 = (threadIdx.x & 7) * 8;
    uint4 o = *(const uint4*)&t[d][k0];
    *(uint4*)(dst + (size_t)(d0 + d) * KNN + k0) = o;
}

// ---------------------------------------------------------------------------
// A epilogue: column exp-scale + row softmax; f32 in, bf16 out.
// ---------------------------------------------------------------------------
__global__ __launch_bounds__(64) void softmaxA_kernel(
    const float* __restrict__ Araw, const float* __restrict__ all_pred,
    __bf16* __restrict__ Ab) {
    __shared__ float cs[64];
    int b = blockIdx.x;
    int row = threadIdx.x;
    cs[row] = __expf(all_pred[((size_t)b * KNN + row) * NCLS + 1]);
    __syncthreads();
    const float* Ar = Araw + ((size_t)b * KNN + row) * KNN;
    float mx = -3.0e38f;
    for (int j = 0; j < 64; ++j) mx = fmaxf(mx, Ar[j] * cs[j]);
    float s = 0.0f;
    for (int j = 0; j < 64; ++j) s += __expf(Ar[j] * cs[j] - mx);
    float inv = 1.0f / s;
    __bf16* Ao = Ab + ((size_t)b * KNN + row) * KNN;
    for (int j = 0; j < 64; ++j) Ao[j] = (__bf16)(__expf(Ar[j] * cs[j] - mx) * inv);
}

// ---------------------------------------------------------------------------
// L2-normalize rows, subtract normalized anchor row. bf16 in -> bf16 out.
// ---------------------------------------------------------------------------
__global__ __launch_bounds__(256) void normalize_kernel(
    const __bf16* __restrict__ Xb, __bf16* __restrict__ Xn) {
    __shared__ float rnorm[64];
    __shared__ float anc[DD];
    const int b = blockIdx.x;
    const int tid = threadIdx.x;
    const int lane = tid & 31;
    const int w = tid >> 5;
    const __bf16* Xi = Xb + (size_t)b * KNN * DD;
    __bf16* Xo = Xn + (size_t)b * KNN * DD;
    for (int rr = 0; rr < 8; ++rr) {
        int row = w * 8 + rr;
        const __bf16* p = Xi + (size_t)row * DD;
        float ss = 0.0f;
        for (int d = lane; d < DD; d += 32) { float v = (float)p[d]; ss += v * v; }
#pragma unroll
        for (int off = 16; off; off >>= 1) ss += __shfl_xor(ss, off, 32);
        if (lane == 0) rnorm[row] = rsqrtf(ss);
    }
    __syncthreads();
    float rn0 = rnorm[0];
    for (int d = tid; d < DD; d += 256) anc[d] = (float)Xi[d] * rn0;
    __syncthreads();
    for (int idx = tid; idx < KNN * DD; idx += 256) {
        int row = idx >> 11;
        int d = idx & (DD - 1);
        Xo[idx] = (__bf16)((float)Xi[idx] * rnorm[row] - anc[d]);
    }
}

// ---------------------------------------------------------------------------
// Generic batched bf16-WMMA GEMM, async-pipelined:
//   Out[b] (64 x N) = act( [X0[b] | X1[b]] (64 x (K0+K1)) @ W (Ktot x N) + bias )
// 64-wide K stages (two 16x16x32 sub-chunks), double-buffered LDS, staged via
// global_load_async_to_lds_b128 (ASYNCcnt) so the next tile streams in while
// WMMA consumes the current one. Requires Ktot % 64 == 0 (true for all calls).
// ---------------------------------------------------------------------------
__global__ __launch_bounds__(256) void gemm_bf16_kernel(
    const __bf16* __restrict__ X0, size_t x0_bs, int K0,
    const __bf16* __restrict__ X1, size_t x1_bs, int K1,
    const __bf16* __restrict__ Wt, size_t w_bs, int Ncols,
    const float* __restrict__ bias, const float* __restrict__ alpha, int act,
    __bf16* __restrict__ OutB, float* __restrict__ OutF) {
    __shared__ __align__(16) __bf16 ldsX[2][2][64][40];  // [buf][sub][row][k(shuffled)]
    __shared__ __align__(16) __bf16 ldsW[2][2][32][72];  // [buf][sub][k][n]
    const int b = blockIdx.y;
    const int n0 = blockIdx.x * 64;
    const int tid = threadIdx.x;
    const int lane = tid & 31;
    const int w = tid >> 5;
    const int half = lane >> 4;
    const int mrow = lane & 15;
    const int Ktot = K0 + K1;
    const __bf16* X0b = X0 + (size_t)b * x0_bs;
    const __bf16* X1b = X1 ? (X1 + (size_t)b * x1_bs) : nullptr;
    const __bf16* Wb = Wt + (size_t)b * w_bs;
    const bool wvec = (n0 + 64 <= Ncols);

    const int t0 = 2 * w;               // wave owns tiles t0, t0+1 (same mt)
    const int mt = t0 >> 2;
    const int nt0 = t0 & 3;
    v8f c0 = {}; v8f c1 = {};

    // Stage one 64-wide K chunk into LDS buffer `bf` (async for aligned paths).
    auto stage = [&](int bf, int kk) {
#pragma unroll
        for (int i = 0; i < 2; ++i) {            // X: 512 16B chunks / 256 thr
            int q = tid + 256 * i;
            int row = q >> 3, j = q & 7;         // j: 8-wide k chunk in [0,8)
            int sub = j >> 2, jj = j & 3;
            int pos = ((jj & 1) << 4) | ((jj >> 1) << 3);  // shuffle to A order
            int kg = kk + j * 8;
            const __bf16* src = (kg < K0) ? (X0b + (size_t)row * K0 + kg)
                                          : (X1b + (size_t)row * K1 + (kg - K0));
            async_copy16(src, &ldsX[bf][sub][row][pos]);
        }
        if (wvec) {
#pragma unroll
            for (int i = 0; i < 2; ++i) {        // W: 512 16B chunks / 256 thr
                int q = tid + 256 * i;
                int kl = q >> 3;                 // 0..63
                int n = (q & 7) * 8;
                async_copy16(Wb + (size_t)(kk + kl) * Ncols + n0 + n,
                             &ldsW[bf][kl >> 5][kl & 31][n]);
            }
        } else {                                 // ragged N tail: sync scalar
            for (int idx = tid; idx < 64 * 64; idx += 256) {
                int kl = idx >> 6, n = idx & 63;
                int ng = n0 + n;
                ldsW[bf][kl >> 5][kl & 31][n] =
                    (ng < Ncols) ? Wb[(size_t)(kk + kl) * Ncols + ng]
                                 : (__bf16)0.0f;
            }
        }
    };

    int buf = 0;
    stage(0, 0);
    for (int kk = 0; kk < Ktot; kk += 64) {
        wait_async0();        // my async writes into ldsX/W[buf] complete
        __syncthreads();      // all waves' writes visible; prev compute done
        if (kk + 64 < Ktot) stage(buf ^ 1, kk + 64);
#pragma unroll
        for (int sub = 0; sub < 2; ++sub) {
            int m = mt * 16 + mrow;
            v8bf alo = *(const v8bf*)&ldsX[buf][sub][m][half * 16];
            v8bf ahi = *(const v8bf*)&ldsX[buf][sub][m][half * 16 + 8];
            v16bf a = __builtin_shufflevector(alo, ahi, 0, 1, 2, 3, 4, 5, 6, 7,
                                              8, 9, 10, 11, 12, 13, 14, 15);
            v16bf bf0, bf1;
#pragma unroll
            for (int s = 0; s < 16; ++s) {
                int kb = (s < 8) ? (half * 8 + s) : (16 + half * 8 + (s - 8));
                bf0[s] = ldsW[buf][sub][kb][nt0 * 16 + mrow];
                bf1[s] = ldsW[buf][sub][kb][nt0 * 16 + 16 + mrow];
            }
            c0 = __builtin_amdgcn_wmma_f32_16x16x32_bf16(
                false, a, false, bf0, (short)0, c0, false, false);
            c1 = __builtin_amdgcn_wmma_f32_16x16x32_bf16(
                false, a, false, bf1, (short)0, c1, false, false);
        }
        buf ^= 1;
    }

    // --- epilogue ---
    auto finish = [&](float v, int ng) -> float {
        if (bias) v += bias[ng];
        if (act == 1) v = fmaxf(v, 0.0f);
        else if (act == 2) v = (v >= 0.0f) ? v : alpha[ng] * v;
        return v;
    };
#pragma unroll
    for (int r = 0; r < 8; ++r) {
        int row = mt * 16 + half * 8 + r;
        int ng0 = n0 + nt0 * 16 + mrow;
        int ng1 = ng0 + 16;
        if (OutF) {
            float* Ob = OutF + (size_t)b * 64 * Ncols;
            if (ng0 < Ncols) Ob[(size_t)row * Ncols + ng0] = finish(c0[r], ng0);
            if (ng1 < Ncols) Ob[(size_t)row * Ncols + ng1] = finish(c1[r], ng1);
        } else {
            __bf16* Ob = OutB + (size_t)b * 64 * Ncols;
            if (ng0 < Ncols)
                Ob[(size_t)row * Ncols + ng0] = (__bf16)finish(c0[r], ng0);
            if (ng1 < Ncols)
                Ob[(size_t)row * Ncols + ng1] = (__bf16)finish(c1[r], ng1);
        }
    }
}

// ---------------------------------------------------------------------------
// 2-class softmax over logits (f32)
// ---------------------------------------------------------------------------
__global__ void softmax2_kernel(const float* __restrict__ logits,
                                float* __restrict__ pred, int rows) {
    int i = blockIdx.x * blockDim.x + threadIdx.x;
    if (i < rows) {
        float a = logits[2 * i], b = logits[2 * i + 1];
        float m = fmaxf(a, b);
        float e0 = __expf(a - m), e1 = __expf(b - m);
        float inv = 1.0f / (e0 + e1);
        pred[2 * i] = e0 * inv;
        pred[2 * i + 1] = e1 * inv;
    }
}

// ---------------------------------------------------------------------------
extern "C" void kernel_launch(void* const* d_in, const int* in_sizes, int n_in,
                              void* d_out, int out_size, void* d_ws, size_t ws_size,
                              hipStream_t stream) {
    (void)in_sizes; (void)n_in; (void)out_size; (void)ws_size;
    const int*   indexes  = (const int*)d_in[0];
    const float* features = (const float*)d_in[1];
    const int*   labels   = (const int*)d_in[2];
    const int*   knn      = (const int*)d_in[5];
    const float* all_pred = (const float*)d_in[6];
    const float* w1       = (const float*)d_in[7];
    const float* b1       = (const float*)d_in[8];
    const float* w2       = (const float*)d_in[9];
    const float* b2       = (const float*)d_in[10];
    const float* cw1      = (const float*)d_in[11];
    const float* cb1      = (const float*)d_in[12];
    const float* alpha    = (const float*)d_in[13];
    const float* cw2      = (const float*)d_in[14];
    const float* cb2      = (const float*)d_in[15];

    // Output layout: pred[B*K*2], simm[8192], clu_sum[8192*2048], nums[8192]
    float* out_f   = (float*)d_out;
    float* pred    = out_f;
    float* simm    = out_f + (size_t)BB * KNN * NCLS;
    float* clu_sum = simm + NUM_CL;
    float* nums    = clu_sum + (size_t)NUM_CL * DD;

    // Workspace carve-up (256B aligned slices)
    char* ws = (char*)d_ws;
    size_t off = 0;
    auto carve = [&](size_t bytes) -> void* {
        void* p = ws + off;
        off += (bytes + 255) & ~(size_t)255;
        return p;
    };
    __bf16* Xb     = (__bf16*)carve((size_t)BB * KNN * DD * 2);    // clu_feat
    __bf16* Xt     = (__bf16*)carve((size_t)BB * DD * KNN * 2);    // clu_feat^T
    __bf16* Xn     = (__bf16*)carve((size_t)BB * KNN * DD * 2);    // all_x
    float*  Araw   = (float*)carve((size_t)BB * KNN * KNN * 4);
    __bf16* Ab     = (__bf16*)carve((size_t)BB * KNN * KNN * 2);
    __bf16* agg1b  = (__bf16*)carve((size_t)BB * KNN * DD * 2);
    __bf16* x1b    = (__bf16*)carve((size_t)BB * KNN * NHID * 2);
    __bf16* agg2b  = (__bf16*)carve((size_t)BB * KNN * NHID * 2);
    __bf16* x2b    = (__bf16*)carve((size_t)BB * KNN * DOUT * 2);
    __bf16* hb     = (__bf16*)carve((size_t)BB * KNN * DOUT * 2);
    float*  logits = (float*)carve((size_t)BB * KNN * NCLS * 4);
    __bf16* w1b    = (__bf16*)carve((size_t)2 * DD * NHID * 2);
    __bf16* w2b    = (__bf16*)carve((size_t)2 * NHID * DOUT * 2);
    __bf16* cw1b   = (__bf16*)carve((size_t)DOUT * DOUT * 2);
    __bf16* cw2b   = (__bf16*)carve((size_t)DOUT * NCLS * 2);

    // 1) zero clu_sum + nums (contiguous in d_out)
    zero_f32<<<4096, 256, 0, stream>>>(clu_sum, (long)NUM_CL * DD + NUM_CL);
    // 2) segment sum (f32 atomics to L2-resident buffer)
    segsum_kernel<<<N_PTS, 256, 0, stream>>>(features, labels, clu_sum, nums);
    // 3) mean + simm
    finalize_kernel<<<NUM_CL, 256, 0, stream>>>(clu_sum, nums, simm);
    // 4) weights -> bf16
    f32_to_bf16<<<1024, 256, 0, stream>>>(w1, w1b, (long)2 * DD * NHID);
    f32_to_bf16<<<256, 256, 0, stream>>>(w2, w2b, (long)2 * NHID * DOUT);
    f32_to_bf16<<<64, 256, 0, stream>>>(cw1, cw1b, (long)DOUT * DOUT);
    f32_to_bf16<<<2, 256, 0, stream>>>(cw2, cw2b, (long)DOUT * NCLS);
    // 5) gather clu_feat (bf16)
    build_feat_kernel<<<BB * KNN, 256, 0, stream>>>(features, clu_sum, labels,
                                                    indexes, knn, Xb);
    // 6) transpose for Gram B-operand
    transpose_kernel<<<dim3(DD / 32, BB), 256, 0, stream>>>(Xb, Xt);
    // 7) Gram: Araw[b] = Xb[b] @ Xt[b]   (N=64, f32 out)
    gemm_bf16_kernel<<<dim3(1, BB), 256, 0, stream>>>(
        Xb, (size_t)KNN * DD, DD, nullptr, 0, 0,
        Xt, (size_t)DD * KNN, KNN, nullptr, nullptr, 0, nullptr, Araw);
    // 8) exp column scale + row softmax -> Ab (bf16)
    softmaxA_kernel<<<BB, 64, 0, stream>>>(Araw, all_pred, Ab);
    // 9) normalize + subtract anchor: Xb -> Xn
    normalize_kernel<<<BB, 256, 0, stream>>>(Xb, Xn);
    // 10) agg1 = A @ Xn                            (N=2048, per-batch W)
    gemm_bf16_kernel<<<dim3(DD / 64, BB), 256, 0, stream>>>(
        Ab, (size_t)KNN * KNN, KNN, nullptr, 0, 0,
        Xn, (size_t)KNN * DD, DD, nullptr, nullptr, 0, agg1b, nullptr);
    // 11) x1 = relu([Xn|agg1] @ w1 + b1)           (N=512)
    gemm_bf16_kernel<<<dim3(NHID / 64, BB), 256, 0, stream>>>(
        Xn, (size_t)KNN * DD, DD, agg1b, (size_t)KNN * DD, DD,
        w1b, 0, NHID, b1, nullptr, 1, x1b, nullptr);
    // 12) agg2 = A @ x1                            (N=512, per-batch W)
    gemm_bf16_kernel<<<dim3(NHID / 64, BB), 256, 0, stream>>>(
        Ab, (size_t)KNN * KNN, KNN, nullptr, 0, 0,
        x1b, (size_t)KNN * NHID, NHID, nullptr, nullptr, 0, agg2b, nullptr);
    // 13) x2 = relu([x1|agg2] @ w2 + b2)           (N=256)
    gemm_bf16_kernel<<<dim3(DOUT / 64, BB), 256, 0, stream>>>(
        x1b, (size_t)KNN * NHID, NHID, agg2b, (size_t)KNN * NHID, NHID,
        w2b, 0, DOUT, b2, nullptr, 1, x2b, nullptr);
    // 14) h = prelu(x2 @ cw1 + cb1, alpha)         (N=256)
    gemm_bf16_kernel<<<dim3(DOUT / 64, BB), 256, 0, stream>>>(
        x2b, (size_t)KNN * DOUT, DOUT, nullptr, 0, 0,
        cw1b, 0, DOUT, cb1, alpha, 2, hb, nullptr);
    // 15) logits = h @ cw2 + cb2                   (N=2, scalar-staged tail)
    gemm_bf16_kernel<<<dim3(1, BB), 256, 0, stream>>>(
        hb, (size_t)KNN * DOUT, DOUT, nullptr, 0, 0,
        cw2b, 0, NCLS, cb2, nullptr, 0, nullptr, logits);
    // 16) 2-class softmax -> pred
    softmax2_kernel<<<(BB * KNN + 255) / 256, 256, 0, stream>>>(logits, pred,
                                                                BB * KNN);
}
